// Attend_60112362275431
// MI455X (gfx1250) — matrix operand order repaired and checked by reading, not compiled, not run on hardware
//
#include <hip/hip_runtime.h>
#include <stdint.h>

// ---------------------------------------------------------------------------
// Flash-attention forward for (B,H,D,N) = (4,16,64,2048), fp32 in/out,
// bf16 WMMA compute on gfx1250 (CDNA5, wave32).
// S^T = K^T Q formulation: each query row lives in one lane (scalar softmax
// stats, no butterflies), O = V P^T lands directly in the (d, n) output
// layout. 64-column KV tiles per barrier, 2x 32-column WMMA passes each.
// ---------------------------------------------------------------------------

typedef __attribute__((ext_vector_type(16))) __bf16 v16bf;
typedef __attribute__((ext_vector_type(8)))  float  v8f;
typedef __attribute__((ext_vector_type(2)))  float  v2f;
typedef __attribute__((ext_vector_type(2)))  __bf16 v2bf;

#define NSEQ     2048
#define DHEAD    64
#define NHEADS   64            // B*H = 4*16
#define BLOCK_M  128           // 8 waves * 16 query rows
#define TJ       64            // KV columns staged per barrier
#define KBS      72            // kb row stride in bf16 (row = j), 16B-multiple
#define VBS      72            // vb row stride in bf16 (row = d), 16B-multiple
#define LOG2E    1.44269504088896340736f
#define SWAPX16  0x21F         // ds_swizzle: or=0, xor=0x10, and=0x1F

static __device__ __forceinline__ uint32_t pk_bf16(float a, float b) {
  v2f f; f.x = a; f.y = b;
  v2bf h = __builtin_convertvector(f, v2bf);   // single v_cvt_pk_bf16_f32
  union { v2bf h; uint32_t u; } c; c.h = h;
  return c.u;
}

static __device__ __forceinline__ float xor16_f(float v) {
  union { float f; int i; } c; c.f = v;
  c.i = __builtin_amdgcn_ds_swizzle(c.i, SWAPX16);
  return c.f;
}
static __device__ __forceinline__ uint32_t xor16_u(uint32_t v) {
  return (uint32_t)__builtin_amdgcn_ds_swizzle((int)v, SWAPX16);
}

union AFrag { v16bf v; uint32_t w[8]; uint4 q[2]; };
union BFrag { v16bf v; uint32_t w[8]; uint4 q[2]; };

__global__ __launch_bounds__(256)
void attend_fa_kernel(const float* __restrict__ qg,
                      const float* __restrict__ kg,
                      const float* __restrict__ vg,
                      float* __restrict__ outg)
{
  __shared__ __align__(16) uint16_t kb[TJ * KBS];       // K tile as (j, d)
  __shared__ __align__(16) uint16_t vb[DHEAD * VBS];    // V tile as (d, j)

  const int bh    = blockIdx.x >> 4;
  const int itile = blockIdx.x & 15;
  const size_t base = (size_t)bh * (DHEAD * NSEQ);
  const int i0 = itile * BLOCK_M;

  const int tid  = threadIdx.x;
  const int wave = tid >> 5;
  const int lane = tid & 31;
  const int hx   = lane >> 4;              // lane half (0/1)
  const int ln   = lane & 15;              // lane-in-half = n index

  // ---- Q tile -> two B fragments (K-dim = d), resident whole kernel ----
  // B layout: n = ln, k = 2r + s + 8*hx (+8 for r >= 4)  (per 32-wide K chunk)
  BFrag bq[2];
  const int qcol = i0 + wave * 16 + ln;    // this lane's query row i
  #pragma unroll
  for (int db = 0; db < 2; ++db) {
    #pragma unroll
    for (int r = 0; r < 8; ++r) {
      const int d = 32 * db + 8 * hx + 2 * r + (r >= 4 ? 8 : 0);
      const float a = __builtin_nontemporal_load(qg + base + (size_t)d * NSEQ + qcol);
      const float b = __builtin_nontemporal_load(qg + base + (size_t)(d + 1) * NSEQ + qcol);
      bq[db].w[r] = pk_bf16(a, b);
    }
  }

  const v8f vzero = {0.f, 0.f, 0.f, 0.f, 0.f, 0.f, 0.f, 0.f};
  v8f oacc[4];                              // O tile: (d = 16t + r + 8*hx, i = ln)
  #pragma unroll
  for (int t = 0; t < 4; ++t) oacc[t] = vzero;

  float rmax = -1e30f, rsum = 0.f;          // per-lane scalars (row i = qcol)
  const float cst = 0.125f * LOG2E;         // D^-0.5 * log2(e)

  // cooperative staging maps (64-wide tile)
  const int kcol = tid & 63;                // K: j column
  const int kd0  = (tid >> 6) * 16;         // K: first of 16 d rows
  const int vrow = tid >> 2;                // V: d row
  const int vc0  = (tid & 3) * 16;          // V: first of 16 j columns

  for (int j0 = 0; j0 < NSEQ; j0 += TJ) {
    __syncthreads();                        // previous tile fully consumed
    // ---- K tile: 16 coalesced b32 loads, two ds_store_b128 ----
    {
      const float* ks = kg + base + (size_t)kd0 * NSEQ + j0 + kcol;
      uint4 w0, w1;
      w0.x = pk_bf16(ks[0 * NSEQ],  ks[1 * NSEQ]);
      w0.y = pk_bf16(ks[2 * NSEQ],  ks[3 * NSEQ]);
      w0.z = pk_bf16(ks[4 * NSEQ],  ks[5 * NSEQ]);
      w0.w = pk_bf16(ks[6 * NSEQ],  ks[7 * NSEQ]);
      w1.x = pk_bf16(ks[8 * NSEQ],  ks[9 * NSEQ]);
      w1.y = pk_bf16(ks[10 * NSEQ], ks[11 * NSEQ]);
      w1.z = pk_bf16(ks[12 * NSEQ], ks[13 * NSEQ]);
      w1.w = pk_bf16(ks[14 * NSEQ], ks[15 * NSEQ]);
      *(uint4*)&kb[kcol * KBS + kd0]     = w0;
      *(uint4*)&kb[kcol * KBS + kd0 + 8] = w1;
      if (j0 + TJ < NSEQ) __builtin_prefetch(ks + TJ, 0, 0);
    }
    // ---- V tile: four float4 loads, two ds_store_b128 ----
    {
      const float* vs = vg + base + (size_t)vrow * NSEQ + j0 + vc0;
      const float4 f0 = *(const float4*)(vs + 0);
      const float4 f1 = *(const float4*)(vs + 4);
      const float4 f2 = *(const float4*)(vs + 8);
      const float4 f3 = *(const float4*)(vs + 12);
      uint4 w0, w1;
      w0.x = pk_bf16(f0.x, f0.y);  w0.y = pk_bf16(f0.z, f0.w);
      w0.z = pk_bf16(f1.x, f1.y);  w0.w = pk_bf16(f1.z, f1.w);
      w1.x = pk_bf16(f2.x, f2.y);  w1.y = pk_bf16(f2.z, f2.w);
      w1.z = pk_bf16(f3.x, f3.y);  w1.w = pk_bf16(f3.z, f3.w);
      *(uint4*)&vb[vrow * VBS + vc0]     = w0;
      *(uint4*)&vb[vrow * VBS + vc0 + 8] = w1;
      if (j0 + TJ < NSEQ) __builtin_prefetch(vs + TJ, 0, 0);
    }
    __syncthreads();

    // ---- two 32-column passes over the staged tile ----
    #pragma unroll
    for (int jj = 0; jj < 2; ++jj) {
      const int jb = jj * 32;

      // S^T tiles: c0 -> j in [jb, jb+16), c1 -> [jb+16, jb+32)
      v8f c0 = vzero, c1 = vzero;
      #pragma unroll
      for (int db = 0; db < 2; ++db) {
        AFrag ak0, ak1;
        const uint16_t* r0 = &kb[(jb + ln) * KBS + 32 * db + 8 * hx];
        const uint16_t* r1 = &kb[(jb + 16 + ln) * KBS + 32 * db + 8 * hx];
        ak0.q[0] = *(const uint4*)r0;  ak0.q[1] = *(const uint4*)(r0 + 16);
        ak1.q[0] = *(const uint4*)r1;  ak1.q[1] = *(const uint4*)(r1 + 16);
        c0 = __builtin_amdgcn_wmma_f32_16x16x32_bf16(false, ak0.v, false, bq[db].v,
                                                     (short)0, c0, false, false);
        c1 = __builtin_amdgcn_wmma_f32_16x16x32_bf16(false, ak1.v, false, bq[db].v,
                                                     (short)0, c1, false, false);
      }

      // online softmax: row i is per-lane; 16 j-values in registers
      float t0 = fmaxf(c0[0], c0[1]), t1 = fmaxf(c0[2], c0[3]);
      float t2 = fmaxf(c0[4], c0[5]), t3 = fmaxf(c0[6], c0[7]);
      float t4 = fmaxf(c1[0], c1[1]), t5 = fmaxf(c1[2], c1[3]);
      float t6 = fmaxf(c1[4], c1[5]), t7 = fmaxf(c1[6], c1[7]);
      t0 = fmaxf(fmaxf(t0, t1), fmaxf(t2, t3));
      t4 = fmaxf(fmaxf(t4, t5), fmaxf(t6, t7));
      float tmax = fmaxf(t0, t4);
      tmax = fmaxf(tmax, xor16_f(tmax));               // other half's 16 j's
      const float nm = fmaxf(rmax, tmax);
      const float fs = exp2f((rmax - nm) * cst);
      rmax = nm;
      const float mneg = -nm * cst;                    // exp arg = fma(c, cst, mneg)

      float p0[8], p1[8];
      #pragma unroll
      for (int r = 0; r < 8; ++r) {
        p0[r] = exp2f(__builtin_fmaf(c0[r], cst, mneg));
        p1[r] = exp2f(__builtin_fmaf(c1[r], cst, mneg));
      }
      float ts = (((p0[0] + p0[1]) + (p0[2] + p0[3])) +
                  ((p0[4] + p0[5]) + (p0[6] + p0[7]))) +
                 (((p1[0] + p1[1]) + (p1[2] + p1[3])) +
                  ((p1[4] + p1[5]) + (p1[6] + p1[7])));
      ts += xor16_f(ts);
      rsum = __builtin_fmaf(rsum, fs, ts);

      if (__ballot(fs < 1.0f) != 0ull) {               // uniform: skip rescale
        #pragma unroll                                 // once running max is set
        for (int t = 0; t < 4; ++t)
          #pragma unroll
          for (int r = 0; r < 8; ++r) oacc[t][r] *= fs;
      }

      // P^T B-fragment (k = j, n = i): send the pack the partner half needs,
      // one xor-16 exchange of 4 dwords.
      uint32_t pk0[4], pk1[4], x[4];
      #pragma unroll
      for (int c = 0; c < 4; ++c) {
        pk0[c] = pk_bf16(p0[2 * c], p0[2 * c + 1]);
        pk1[c] = pk_bf16(p1[2 * c], p1[2 * c + 1]);
      }
      #pragma unroll
      for (int c = 0; c < 4; ++c) x[c] = xor16_u(hx ? pk0[c] : pk1[c]);
      BFrag pf;
      #pragma unroll
      for (int c = 0; c < 4; ++c) {
        pf.w[c]     = hx ? x[c]   : pk0[c];            // j = 2c+s + 16*hx
        pf.w[4 + c] = hx ? pk1[c] : x[c];              // j = 8+2c+s + 16*hx
      }

      // O += V @ P^T : A = V tile (m = d, k = j), 4 d-subtiles
      #pragma unroll
      for (int t = 0; t < 4; ++t) {
        AFrag av;
        const uint16_t* r0 = &vb[(16 * t + ln) * VBS + jb + 16 * hx];
        av.q[0] = *(const uint4*)r0;
        av.q[1] = *(const uint4*)(r0 + 8);
        oacc[t] = __builtin_amdgcn_wmma_f32_16x16x32_bf16(false, av.v, false, pf.v,
                                                          (short)0, oacc[t], false, false);
      }
    }
  }

  // ---- normalize and store: C layout is already (d, i) -> coalesced runs ----
  const float inv = 1.0f / rsum;
  #pragma unroll
  for (int t = 0; t < 4; ++t) {
    #pragma unroll
    for (int r = 0; r < 8; ++r) {
      const int d = 16 * t + r + 8 * hx;
      __builtin_nontemporal_store(oacc[t][r] * inv,
                                  outg + base + (size_t)d * NSEQ + qcol);
    }
  }
}

extern "C" void kernel_launch(void* const* d_in, const int* in_sizes, int n_in,
                              void* d_out, int out_size, void* d_ws, size_t ws_size,
                              hipStream_t stream) {
  (void)in_sizes; (void)n_in; (void)out_size; (void)d_ws; (void)ws_size;
  const float* q = (const float*)d_in[0];
  const float* k = (const float*)d_in[1];
  const float* v = (const float*)d_in[2];
  float* out = (float*)d_out;
  const int grid = NHEADS * (NSEQ / BLOCK_M);   // 64 * 16 = 1024 blocks
  attend_fa_kernel<<<grid, 256, 0, stream>>>(q, k, v, out);
}
